// ClusterRetrieval_515396076312
// MI455X (gfx1250) — matrix operand order repaired and checked by reading, not compiled, not run on hardware
//
#include <hip/hip_runtime.h>
#include <hip/hip_bf16.h>
#include <stdint.h>

typedef __attribute__((ext_vector_type(16))) __bf16 v16bf;
typedef __attribute__((ext_vector_type(8)))  __bf16 v8bf;
typedef __attribute__((ext_vector_type(8)))  float  v8f;

#define DDIM 128
#define BQ   16384            // queries (fixed by reference)
#define NC   16384            // centroids (fixed by reference)
#define BROW_PAD 136          // bf16 elements per padded row (128 + 8 pad)
#define BTILE_ELEMS (16 * BROW_PAD)      // one padded 16x128 tile (2176 elems)
#define CTILE_BYTES (2 * BTILE_ELEMS * 2) // hi tile + lo tile = 8704 bytes

// ---------------------------------------------------------------------------
// Kernel 1: split f32 centroids into bf16 hi/lo, PRE-PADDED tile layout that
// exactly matches the LDS staging layout: per 16-centroid tile, a contiguous
// 8704B block = [16 padded hi rows][16 padded lo rows], row stride 272B.
// This lets the async global->LDS copies share one immediate offset for both
// the LDS and global address (ISA: INST_OFFSET is added to both).
// ---------------------------------------------------------------------------
__global__ void split_bf16_kernel(const float* __restrict__ src,
                                  __bf16* __restrict__ dst, int n) {
    int i = blockIdx.x * blockDim.x + threadIdx.x;
    int stride = gridDim.x * blockDim.x;
    for (; i < n; i += stride) {
        float v = __builtin_nontemporal_load(src + i);
        __bf16 h = (__bf16)v;
        __bf16 l = (__bf16)(v - (float)h);
        int row = i >> 7;            // centroid index
        int k   = i & 127;           // dim index
        int tile = row >> 4;
        int r    = row & 15;
        size_t base = (size_t)tile * (2 * BTILE_ELEMS) + r * BROW_PAD + k;
        dst[base] = h;
        dst[base + BTILE_ELEMS] = l;
    }
}

// One async b128 copy, LDS and global advanced by the same immediate offset.
template <int OFF>
__device__ __forceinline__ void async_ld_b128(uint32_t lds, uint64_t gaddr) {
    asm volatile("global_load_async_to_lds_b128 %0, %1, off offset:%c2"
                 :: "v"(lds), "v"(gaddr), "n"(OFF) : "memory");
}

// Branchless insert into a sorted-descending top-8 list held in registers.
__device__ __forceinline__ void topk_insert8(float (&tv)[8], int (&ti)[8],
                                             float v, int id) {
    float nv = v; int ni = id;
#pragma unroll
    for (int i = 0; i < 8; ++i) {
        bool gt = nv > tv[i];
        float ov = tv[i]; int oi = ti[i];
        tv[i] = gt ? nv : ov;  ti[i] = gt ? ni : oi;
        nv   = gt ? ov : nv;   ni   = gt ? oi : ni;
    }
}

// ---------------------------------------------------------------------------
// Kernel 2: fused bf16x3 WMMA GEMM + all_sims store + per-row top-8.
// Block = 128 threads (4 waves), owns 16 query rows. Wave w handles cluster
// tiles j = w, w+4, w+8, ... with double-buffered async global->LDS staging.
// ---------------------------------------------------------------------------
__launch_bounds__(128)
__global__ void cluster_topk_kernel(const float* __restrict__ query,
                                    const __bf16* __restrict__ cdata,
                                    float* __restrict__ out) {
    __shared__ __align__(64) __bf16 q_hi[16 * DDIM];
    __shared__ __align__(64) __bf16 q_lo[16 * DDIM];
    // [wave][buf][hi/lo][padded 16x128 tile] -- [hi][lo] contiguous per buf
    __shared__ __align__(64) __bf16 btile[4][2][2][BTILE_ELEMS];
    __shared__ __align__(64) float  dtile[4][16][20];       // padded 16x16
    __shared__ float cand_val[4][16][2][8];
    __shared__ int   cand_idx[4][16][2][8];

    const int tid  = threadIdx.x;
    const int wave = tid >> 5;
    const int lane = tid & 31;
    const int m0   = blockIdx.x * 16;

    float* ids_out = out;                                   // B*8 floats
    float* val_out = out + (size_t)BQ * 8;                  // B*8 floats
    float* all_out = out + (size_t)BQ * 16;                 // B*N floats

    // ---- split query tile into bf16 hi/lo in LDS (contiguous 2048 elems)
    const float* gq = query + (size_t)m0 * DDIM;
    for (int e = tid; e < 16 * DDIM; e += 128) {
        float v = gq[e];
        __bf16 h = (__bf16)v;
        q_hi[e] = h;
        q_lo[e] = (__bf16)(v - (float)h);
    }
    __syncthreads();

    // ---- build resident A fragments (WMMA 16-bit A layout)
    // lanes 0-15: row=lane, K groups {0..7, 16..23}; lanes 16-31: row=lane-16,
    // K groups {8..15, 24..31} (per 32-K chunk).
    const int mrow = lane & 15;
    const int aoff = (lane < 16) ? 0 : 8;
    v16bf a_hi[4], a_lo[4];
#pragma unroll
    for (int kc = 0; kc < 4; ++kc) {
        int b1 = mrow * DDIM + kc * 32 + aoff;
        v8bf h0 = *(const v8bf*)&q_hi[b1];
        v8bf h1 = *(const v8bf*)&q_hi[b1 + 16];
        v8bf l0 = *(const v8bf*)&q_lo[b1];
        v8bf l1 = *(const v8bf*)&q_lo[b1 + 16];
        a_hi[kc] = __builtin_shufflevector(h0, h1, 0,1,2,3,4,5,6,7,8,9,10,11,12,13,14,15);
        a_lo[kc] = __builtin_shufflevector(l0, l1, 0,1,2,3,4,5,6,7,8,9,10,11,12,13,14,15);
    }

    // ---- per-lane copy slot: row pair (lane>>4)+2i, 16B column chunk
    const uint32_t lrow = (uint32_t)(lane >> 4) * (BROW_PAD * 2)
                        + (uint32_t)(lane & 15) * 16u;
    uint32_t lds_l0[2];
    lds_l0[0] = (uint32_t)(uintptr_t)&btile[wave][0][0][0] + lrow;
    lds_l0[1] = (uint32_t)(uintptr_t)&btile[wave][1][0][0] + lrow;
    const uint64_t cbytes = (uint64_t)(uintptr_t)cdata + lrow;

    // 16 async b128 copies (8KB tile: hi 4352B + lo 4352B, identical global
    // and LDS padded layout -> single shared immediate offset per copy).
    auto prefetch = [&](int j, uint32_t l0) {
        uint64_t g0 = cbytes + (uint64_t)((uint32_t)j * (uint32_t)CTILE_BYTES);
        async_ld_b128<0 * 544>(l0, g0);
        async_ld_b128<1 * 544>(l0, g0);
        async_ld_b128<2 * 544>(l0, g0);
        async_ld_b128<3 * 544>(l0, g0);
        async_ld_b128<4 * 544>(l0, g0);
        async_ld_b128<5 * 544>(l0, g0);
        async_ld_b128<6 * 544>(l0, g0);
        async_ld_b128<7 * 544>(l0, g0);
        async_ld_b128<8 * 544>(l0, g0);
        async_ld_b128<9 * 544>(l0, g0);
        async_ld_b128<10 * 544>(l0, g0);
        async_ld_b128<11 * 544>(l0, g0);
        async_ld_b128<12 * 544>(l0, g0);
        async_ld_b128<13 * 544>(l0, g0);
        async_ld_b128<14 * 544>(l0, g0);
        async_ld_b128<15 * 544>(l0, g0);
    };

    // ---- per-lane top-8 state: row r = lane%16, column-half h = lane/16
    const int r = lane & 15;
    const int h = lane >> 4;
    float tv[8]; int ti[8];
#pragma unroll
    for (int i = 0; i < 8; ++i) { tv[i] = -1e30f; ti[i] = -1; }

    const int ITER = NC / 64;  // tiles per wave (N/16 tiles, 4 waves)
    prefetch(wave, lds_l0[0]);

    const int nlocal = lane & 15;
    const int boff   = (lane < 16) ? 0 : 8;
    const int hi8    = (lane < 16) ? 0 : 8;

    // Uniform store base + per-lane 32-bit element offset; per-row deltas
    // v*NC fold into the 24-bit instruction offset of global_store.
    float* const sbase = all_out + (size_t)m0 * NC;

    for (int it = 0; it < ITER; ++it) {
        const int j   = wave + it * 4;
        const int buf = it & 1;
        if (it + 1 < ITER) {
            prefetch(j + 4, lds_l0[buf ^ 1]);
            asm volatile("s_wait_asynccnt 0x10" ::: "memory");  // prev batch done
        } else {
            asm volatile("s_wait_asynccnt 0x0" ::: "memory");
        }

        const __bf16* bh_base = &btile[wave][buf][0][0];
        const __bf16* bl_base = &btile[wave][buf][1][0];

        // Two accumulators: halve the serial WMMA RAW chain (6 deep each).
        v8f acc0 = {};
        v8f acc1 = {};
#pragma unroll
        for (int kc = 0; kc < 4; ++kc) {
            int be = nlocal * BROW_PAD + kc * 32 + boff;  // 16B-aligned
            v8bf bh0 = *(const v8bf*)&bh_base[be];
            v8bf bh1 = *(const v8bf*)&bh_base[be + 16];
            v8bf bl0 = *(const v8bf*)&bl_base[be];
            v8bf bl1 = *(const v8bf*)&bl_base[be + 16];
            v16bf bh = __builtin_shufflevector(bh0, bh1, 0,1,2,3,4,5,6,7,8,9,10,11,12,13,14,15);
            v16bf bl = __builtin_shufflevector(bl0, bl1, 0,1,2,3,4,5,6,7,8,9,10,11,12,13,14,15);
            acc0 = __builtin_amdgcn_wmma_f32_16x16x32_bf16(false, a_hi[kc], false, bh,
                                                           (short)0, acc0, false, false);
            acc1 = __builtin_amdgcn_wmma_f32_16x16x32_bf16(false, a_hi[kc], false, bl,
                                                           (short)0, acc1, false, false);
            if (kc & 1)
                acc1 = __builtin_amdgcn_wmma_f32_16x16x32_bf16(false, a_lo[kc], false, bh,
                                                               (short)0, acc1, false, false);
            else
                acc0 = __builtin_amdgcn_wmma_f32_16x16x32_bf16(false, a_lo[kc], false, bh,
                                                               (short)0, acc0, false, false);
        }
        v8f acc = acc0 + acc1;

        // ---- store 16x16 D tile to all_similarities (saddr-base + imm,
        // non-temporal write-once stream) and stage into padded LDS for the
        // row-wise top-k scan.
        const uint32_t voff = (uint32_t)hi8 * NC + (uint32_t)j * 16 + (uint32_t)nlocal;
#pragma unroll
        for (int v = 0; v < 8; ++v) {
            __builtin_nontemporal_store(acc[v], sbase + voff + (uint32_t)v * NC);
            dtile[wave][v + hi8][nlocal] = acc[v];
        }
        __builtin_amdgcn_wave_barrier();

        // Row-wise scan: 8 contiguous floats -> two ds_load_b128.
        const float4* dp = (const float4*)&dtile[wave][r][h * 8];
        float4 f0 = dp[0];
        float4 f1 = dp[1];
        const int id0 = j * 16 + h * 8;
        if (f0.x > tv[7]) topk_insert8(tv, ti, f0.x, id0 + 0);
        if (f0.y > tv[7]) topk_insert8(tv, ti, f0.y, id0 + 1);
        if (f0.z > tv[7]) topk_insert8(tv, ti, f0.z, id0 + 2);
        if (f0.w > tv[7]) topk_insert8(tv, ti, f0.w, id0 + 3);
        if (f1.x > tv[7]) topk_insert8(tv, ti, f1.x, id0 + 4);
        if (f1.y > tv[7]) topk_insert8(tv, ti, f1.y, id0 + 5);
        if (f1.z > tv[7]) topk_insert8(tv, ti, f1.z, id0 + 6);
        if (f1.w > tv[7]) topk_insert8(tv, ti, f1.w, id0 + 7);
        __builtin_amdgcn_wave_barrier();
    }

    // ---- dump per-lane candidates; merge 64 -> 8 per row
#pragma unroll
    for (int i = 0; i < 8; ++i) {
        cand_val[wave][r][h][i] = tv[i];
        cand_idx[wave][r][h][i] = ti[i];
    }
    __syncthreads();

    if (tid < 16) {
        float fv[8]; int fi[8];
#pragma unroll
        for (int i = 0; i < 8; ++i) { fv[i] = -1e30f; fi[i] = -1; }
        for (int w = 0; w < 4; ++w)
            for (int hh = 0; hh < 2; ++hh)
#pragma unroll
                for (int i = 0; i < 8; ++i) {
                    float v = cand_val[w][tid][hh][i];
                    if (v > fv[7]) topk_insert8(fv, fi, v, cand_idx[w][tid][hh][i]);
                }
        const size_t m = (size_t)m0 + tid;
#pragma unroll
        for (int i = 0; i < 8; ++i) {
            ids_out[m * 8 + i] = (fv[i] >= 0.0f) ? (float)fi[i] : -1.0f;
            val_out[m * 8 + i] = fv[i];
        }
    }
}

// ---------------------------------------------------------------------------
extern "C" void kernel_launch(void* const* d_in, const int* in_sizes, int n_in,
                              void* d_out, int out_size, void* d_ws, size_t ws_size,
                              hipStream_t stream) {
    const float* query    = (const float*)d_in[0];
    const float* clusters = (const float*)d_in[1];
    (void)n_in; (void)out_size; (void)ws_size; (void)in_sizes;

    __bf16* cdata = (__bf16*)d_ws;   // (NC/16) tiles * 8704B = ~8.9 MB

    split_bf16_kernel<<<512, 256, 0, stream>>>(clusters, cdata, NC * DDIM);
    cluster_topk_kernel<<<BQ / 16, 128, 0, stream>>>(query, cdata, (float*)d_out);
}
